// Encoder_50294067036840
// MI455X (gfx1250) — matrix operand order repaired and checked by reading, not compiled, not run on hardware
//
#include <hip/hip_runtime.h>
#include <hip/hip_bf16.h>

#define FD 128          // feature dim
#define LDS_STRIDE 136  // padded row stride (elements) -> conflict-free b128 reads
#define BN_EPS 1e-5f

typedef __attribute__((ext_vector_type(16))) __bf16 v16bf;
typedef __attribute__((ext_vector_type(8)))  __bf16 v8bf;
typedef __attribute__((ext_vector_type(8)))  float  v8f;

// Permute k within each 32-wide chunk so a lane's 16 fragment elements are
// contiguous: element j of (half) lives at chunk-offset half*16 + j.
__device__ __forceinline__ int perm_k(int k) {
    return (k & ~31) + (((k >> 3) & 1) << 4) + (k & 7) + (((k >> 4) & 1) << 3);
}

// ---------------------------------------------------------------------------
// GEMM: H[n, :] = X[n, :] @ W   (N x 128) * (128 x 128), bf16 WMMA, f32 accum
// Block = 256 threads = 8 waves; block covers 128 rows; wave covers 16 rows.
// A tile staged in natural k order, W staged transposed -> fragments are 2x b128 loads.
// ---------------------------------------------------------------------------
__global__ __launch_bounds__(256) void gemm_bf16_kernel(
    const float* __restrict__ X, const float* __restrict__ W,
    float* __restrict__ H, int nrows)
{
    __shared__ __bf16 sXp[FD * LDS_STRIDE];   // ~34 KB, X tile, natural k order
    __shared__ __bf16 sWT[FD * LDS_STRIDE];   // ~34 KB, transposed W
    const int tid  = threadIdx.x;
    const int row0 = blockIdx.x * FD;

    // Stage X tile: float4 along k, convert to bf16, write 8B at same k pos
    for (int i = tid * 4; i < FD * FD; i += 256 * 4) {
        const int r = i >> 7, c = i & 127;
        int gr = row0 + r; if (gr >= nrows) gr = nrows - 1;
        const float4 x4 = *(const float4*)(X + (size_t)gr * FD + c);
        __bf16 t[4] = { (__bf16)x4.x, (__bf16)x4.y, (__bf16)x4.z, (__bf16)x4.w };
        *(uint2*)&sXp[r * LDS_STRIDE + c] = *(const uint2*)t;
    }
    // Stage W transposed: sWT[n][k].  Per pass: 128 consecutive n for a fixed
    // k row -> coalesced global reads; 8B contiguous LDS store per thread.
    for (int p = tid; p < (FD / 4) * FD; p += 256) {
        const int n = p & 127;
        const int k0 = (p >> 7) * 4;
        __bf16 t[4];
        #pragma unroll
        for (int i = 0; i < 4; ++i)
            t[i] = (__bf16)W[(size_t)(k0 + i) * FD + n];
        *(uint2*)&sWT[n * LDS_STRIDE + k0] = *(const uint2*)t;
    }
    __syncthreads();

    const int wave = tid >> 5, lane = tid & 31;
    const int half = lane >> 4, l15 = lane & 15;
    const int mrow = wave * 16 + l15;          // A-matrix row for this lane

    v8f acc[8] = {};                           // 8 column tiles of 16

    #pragma unroll
    for (int kc = 0; kc < 4; ++kc) {           // K = 128 in chunks of 32
        const __bf16* ap = &sXp[mrow * LDS_STRIDE + kc * 32 + half * 16];
        const v8bf alo = *(const v8bf*)ap;
        const v8bf ahi = *(const v8bf*)(ap + 8);
        const v16bf a = __builtin_shufflevector(alo, ahi,
            0, 1, 2, 3, 4, 5, 6, 7, 8, 9, 10, 11, 12, 13, 14, 15);
        #pragma unroll
        for (int ct = 0; ct < 8; ++ct) {
            const __bf16* bp =
                &sWT[(ct * 16 + l15) * LDS_STRIDE + kc * 32 + half * 16];
            const v8bf blo = *(const v8bf*)bp;
            const v8bf bhi = *(const v8bf*)(bp + 8);
            const v16bf b = __builtin_shufflevector(blo, bhi,
                0, 1, 2, 3, 4, 5, 6, 7, 8, 9, 10, 11, 12, 13, 14, 15);
            acc[ct] = __builtin_amdgcn_wmma_f32_16x16x32_bf16(
                false, a, false, b, (short)0, acc[ct], false, false);
        }
    }

    // D layout: lane holds col n=l15; VGPR r -> row (half*8 + r)
    const int mbase = row0 + wave * 16 + half * 8;
    #pragma unroll
    for (int ct = 0; ct < 8; ++ct) {
        const int col = ct * 16 + l15;
        #pragma unroll
        for (int r = 0; r < 8; ++r) {
            const int gm = mbase + r;
            if (gm < nrows) H[(size_t)gm * FD + col] = acc[ct][r];
        }
    }
}

// ---------------------------------------------------------------------------
// Degree / normalization
// ---------------------------------------------------------------------------
__global__ void deg_init_kernel(float* deg, int n) {
    int i = blockIdx.x * blockDim.x + threadIdx.x;
    if (i < n) deg[i] = 1.0f;                  // self-loop
}
__global__ void deg_accum_kernel(float* deg, const int* __restrict__ dst, int e) {
    int i = blockIdx.x * blockDim.x + threadIdx.x;
    if (i < e) atomicAdd(&deg[dst[i]], 1.0f);
}
__global__ void dinv_kernel(float* deg, int n) {
    int i = blockIdx.x * blockDim.x + threadIdx.x;
    if (i < n) deg[i] = rsqrtf(deg[i]);        // deg >= 1 always
}

// ---------------------------------------------------------------------------
// AGG init: self-loop contribution + bias   (one thread per float4 chunk)
// ---------------------------------------------------------------------------
__global__ void init_agg_kernel(const float* __restrict__ H,
                                const float* __restrict__ dinv,
                                const float* __restrict__ bias,
                                float* __restrict__ AGG, int nrows)
{
    int idx = blockIdx.x * blockDim.x + threadIdx.x;
    if (idx >= nrows * (FD / 4)) return;
    int node = idx >> 5;
    int c = (idx & 31) * 4;
    float di = dinv[node];
    float w = di * di;
    const float4 h4 = *(const float4*)(H + (size_t)node * FD + c);
    const float4 b4 = *(const float4*)(bias + c);
    float4 o;
    o.x = h4.x * w + b4.x; o.y = h4.y * w + b4.y;
    o.z = h4.z * w + b4.z; o.w = h4.w * w + b4.w;
    *(float4*)(AGG + (size_t)node * FD + c) = o;
}

// ---------------------------------------------------------------------------
// Edge scatter: AGG[dst] += H[src] * dinv[src]*dinv[dst]
// One wave (32 lanes) per edge, float4 per lane, 4 f32 atomics.
// ---------------------------------------------------------------------------
__global__ __launch_bounds__(256) void scatter_kernel(
    const float* __restrict__ H, float* __restrict__ AGG,
    const int* __restrict__ src, const int* __restrict__ dst,
    const float* __restrict__ dinv, int nedges)
{
    int e = blockIdx.x * 8 + (threadIdx.x >> 5);
    if (e >= nedges) return;
    int lane = threadIdx.x & 31;
    int s = src[e], d = dst[e];
    float w = dinv[s] * dinv[d];
    const float4 v = *(const float4*)(H + (size_t)s * FD + lane * 4);
    float* p = AGG + (size_t)d * FD + lane * 4;
    atomicAdd(p + 0, v.x * w);
    atomicAdd(p + 1, v.y * w);
    atomicAdd(p + 2, v.z * w);
    atomicAdd(p + 3, v.w * w);
}

// ---------------------------------------------------------------------------
// BatchNorm statistics: per-feature sum and sum-of-squares
// ---------------------------------------------------------------------------
__global__ void zero_sums_kernel(float* sums) { sums[threadIdx.x] = 0.0f; }

__global__ __launch_bounds__(256) void bn_stats_kernel(
    const float* __restrict__ AGG, float* __restrict__ sums,
    int nrows, int rows_per_block)
{
    const int f = threadIdx.x & 127;
    const int rsub = threadIdx.x >> 7;         // 0 or 1
    const int r0 = blockIdx.x * rows_per_block;
    int r1 = r0 + rows_per_block; if (r1 > nrows) r1 = nrows;
    float s = 0.0f, sq = 0.0f;
    for (int r = r0 + rsub; r < r1; r += 2) {
        float v = AGG[(size_t)r * FD + f];
        s += v; sq += v * v;
    }
    __shared__ float red[512];
    red[threadIdx.x] = s;
    red[256 + threadIdx.x] = sq;
    __syncthreads();
    if (threadIdx.x < 128) {
        s  = red[threadIdx.x] + red[threadIdx.x + 128];
        sq = red[256 + threadIdx.x] + red[256 + threadIdx.x + 128];
        atomicAdd(&sums[f], s);
        atomicAdd(&sums[128 + f], sq);
    }
}

__global__ void bn_finalize_kernel(const float* __restrict__ sums,
                                   const float* __restrict__ g,
                                   const float* __restrict__ be,
                                   float* __restrict__ stats, int nrows)
{
    int f = threadIdx.x;                       // 128 threads
    float inv_n = 1.0f / (float)nrows;
    float mu  = sums[f] * inv_n;
    float var = sums[128 + f] * inv_n - mu * mu;
    float rstd = rsqrtf(var + BN_EPS);
    float sc = rstd * g[f];
    stats[f] = sc;                             // scale
    stats[128 + f] = be[f] - mu * sc;          // shift
}

// ---------------------------------------------------------------------------
// BN apply + PReLU
// ---------------------------------------------------------------------------
__global__ void bn_apply_kernel(const float* __restrict__ AGG,
                                const float* __restrict__ stats,
                                const float* __restrict__ aw,
                                float* __restrict__ OUT, int nrows)
{
    int idx = blockIdx.x * blockDim.x + threadIdx.x;
    if (idx >= nrows * (FD / 4)) return;
    int node = idx >> 5;
    int c = (idx & 31) * 4;
    float a = aw[0];
    const float4 v  = *(const float4*)(AGG + (size_t)node * FD + c);
    const float4 sc = *(const float4*)(stats + c);
    const float4 sh = *(const float4*)(stats + 128 + c);
    float4 o;
    float t;
    t = v.x * sc.x + sh.x; o.x = (t >= 0.0f) ? t : a * t;
    t = v.y * sc.y + sh.y; o.y = (t >= 0.0f) ? t : a * t;
    t = v.z * sc.z + sh.z; o.z = (t >= 0.0f) ? t : a * t;
    t = v.w * sc.w + sh.w; o.w = (t >= 0.0f) ? t : a * t;
    *(float4*)(OUT + (size_t)node * FD + c) = o;
}

// ---------------------------------------------------------------------------
// Host-side orchestration
// ---------------------------------------------------------------------------
static void run_layer(const float* Xin, const float* W, const float* b,
                      const float* g, const float* be, const float* aw,
                      const int* src, const int* dst, const float* dinv,
                      float* Hbuf, float* Abuf, float* sums, float* stats,
                      float* OUT, int N, int E, hipStream_t stream)
{
    const int gemm_blocks = (N + FD - 1) / FD;
    gemm_bf16_kernel<<<gemm_blocks, 256, 0, stream>>>(Xin, W, Hbuf, N);

    const int elem4 = N * (FD / 4);
    init_agg_kernel<<<(elem4 + 255) / 256, 256, 0, stream>>>(Hbuf, dinv, b, Abuf, N);

    scatter_kernel<<<(E + 7) / 8, 256, 0, stream>>>(Hbuf, Abuf, src, dst, dinv, E);

    zero_sums_kernel<<<1, 256, 0, stream>>>(sums);
    const int stat_blocks = 128;
    const int rows_per_block = (N + stat_blocks - 1) / stat_blocks;
    bn_stats_kernel<<<stat_blocks, 256, 0, stream>>>(Abuf, sums, N, rows_per_block);
    bn_finalize_kernel<<<1, 128, 0, stream>>>(sums, g, be, stats, N);

    bn_apply_kernel<<<(elem4 + 255) / 256, 256, 0, stream>>>(Abuf, stats, aw, OUT, N);
}

extern "C" void kernel_launch(void* const* d_in, const int* in_sizes, int n_in,
                              void* d_out, int out_size, void* d_ws, size_t ws_size,
                              hipStream_t stream)
{
    const float* x   = (const float*)d_in[0];
    const int*   ei  = (const int*)d_in[1];
    const float* W1  = (const float*)d_in[2];
    const float* b1  = (const float*)d_in[3];
    const float* g1  = (const float*)d_in[4];
    const float* be1 = (const float*)d_in[5];
    const float* a1  = (const float*)d_in[6];
    const float* W2  = (const float*)d_in[7];
    const float* b2  = (const float*)d_in[8];
    const float* g2  = (const float*)d_in[9];
    const float* be2 = (const float*)d_in[10];
    const float* a2  = (const float*)d_in[11];

    const int N = in_sizes[0] / FD;
    const int E = in_sizes[1] / 2;
    const int* src = ei;
    const int* dst = ei + E;

    // Workspace carve-up
    char* ws = (char*)d_ws;
    size_t off = 0;
    float* dinv = (float*)(ws + off); off += ((size_t)N * 4 + 255) & ~(size_t)255;
    float* sums  = (float*)(ws + off); off += 1024;   // 256 floats
    float* stats = (float*)(ws + off); off += 1024;   // 256 floats
    float* bufA = (float*)(ws + off); off += (size_t)N * FD * 4;
    float* bufB = (float*)(ws + off);

    // Symmetric normalization weights
    deg_init_kernel<<<(N + 255) / 256, 256, 0, stream>>>(dinv, N);
    deg_accum_kernel<<<(E + 255) / 256, 256, 0, stream>>>(dinv, dst, E);
    dinv_kernel<<<(N + 255) / 256, 256, 0, stream>>>(dinv, N);

    // Layer 1: x -> bufA(h1) -> bufB(agg1) -> bufA(x2)
    run_layer(x, W1, b1, g1, be1, a1, src, dst, dinv,
              bufA, bufB, sums, stats, bufA, N, E, stream);

    // Layer 2: bufA(x2) -> bufB(h2) -> bufA(agg2) -> d_out
    run_layer(bufA, W2, b2, g2, be2, a2, src, dst, dinv,
              bufB, bufA, sums, stats, (float*)d_out, N, E, stream);
}
